// PerceiverAttention_15298673508818
// MI455X (gfx1250) — matrix operand order repaired and checked by reading, not compiled, not run on hardware
//
#include <hip/hip_runtime.h>
#include <hip/hip_bf16.h>

typedef __bf16 bf16;
typedef __attribute__((ext_vector_type(16))) __bf16 v16bf;
typedef __attribute__((ext_vector_type(8)))  __bf16 bf16x8;
typedef __attribute__((ext_vector_type(8)))  float  v8f;

union BF16V { v16bf v; bf16x8 h[2]; };

// Async global->LDS 16-byte copy (CDNA5, tracked by ASYNCcnt). lds_off is the
// byte offset of the destination within the workgroup LDS allocation.
__device__ __forceinline__ void async_copy_b128(unsigned lds_off, const void* gaddr) {
  asm volatile("global_load_async_to_lds_b128 %0, %1, off"
               :: "v"(lds_off), "v"(gaddr) : "memory");
}
__device__ __forceinline__ unsigned lds_off_of(const void* p) {
  return (unsigned)(size_t)p;  // low 32 bits of a shared-aperture address = LDS offset
}

// ---------------------------------------------------------------------------
// LayerNorm: one 256-thread block per row, fp32 in -> bf16 out.
// ---------------------------------------------------------------------------
template <int ROWLEN>
__global__ __launch_bounds__(256) void ln_kernel(const float* __restrict__ in,
                                                 const float* __restrict__ w,
                                                 const float* __restrict__ b,
                                                 bf16* __restrict__ out) {
  __shared__ float red[8][2];
  const int row = blockIdx.x;
  const float* x = in + (size_t)row * ROWLEN;
  bf16* o = out + (size_t)row * ROWLEN;

  float s = 0.f, ss = 0.f;
  for (int i = threadIdx.x; i < ROWLEN; i += 256) {
    float v = x[i];
    s += v;
    ss += v * v;
  }
  for (int m = 16; m >= 1; m >>= 1) {
    s += __shfl_xor(s, m, 32);
    ss += __shfl_xor(ss, m, 32);
  }
  if ((threadIdx.x & 31) == 0) {
    red[threadIdx.x >> 5][0] = s;
    red[threadIdx.x >> 5][1] = ss;
  }
  __syncthreads();
  s = 0.f; ss = 0.f;
  for (int i = 0; i < 8; i++) { s += red[i][0]; ss += red[i][1]; }
  const float mu = s * (1.0f / ROWLEN);
  const float var = ss * (1.0f / ROWLEN) - mu * mu;
  const float rstd = rsqrtf(var + 1e-5f);
  for (int i = threadIdx.x; i < ROWLEN; i += 256) {
    float v = (x[i] - mu) * rstd * w[i] + b[i];
    o[i] = (bf16)v;
  }
}

// ---------------------------------------------------------------------------
// Transpose + fp32 -> bf16 convert:  in (R x C) row-major  ->  out (C x R).
// ---------------------------------------------------------------------------
__global__ __launch_bounds__(256) void tconv_kernel(const float* __restrict__ in,
                                                    bf16* __restrict__ out,
                                                    int R, int C) {
  int idx = blockIdx.x * 256 + threadIdx.x;
  if (idx >= R * C) return;
  int r = idx / C, c = idx - r * C;
  out[(size_t)c * R + r] = (bf16)in[idx];
}

// ---------------------------------------------------------------------------
// bf16 WMMA GEMM:  C(MxN) = A(MxK) * B(KxN), B given transposed as Bt(NxK).
// 256 threads = 8 waves (2x4), wave = 32x32 C block, WG tile 64x128, BK=32.
// Double-buffered LDS, tiles staged with global_load_async_to_lds_b128 and
// pipelined against the WMMAs via s_wait_asynccnt.
// ---------------------------------------------------------------------------
template <typename OutT>
__global__ __launch_bounds__(256) void gemm_bf16(const bf16* __restrict__ A,
                                                 const bf16* __restrict__ Bt,
                                                 OutT* __restrict__ C,
                                                 int M, int K, int N) {
  constexpr int BM = 64, BN = 128, BK = 32;
  constexpr int AS = 48, BS = 48;  // padded LDS strides (x2B = 96B, 16B mult)
  __shared__ __align__(16) bf16 sA[2][BM * AS];
  __shared__ __align__(16) bf16 sB[2][BN * BS];

  const int m0 = blockIdx.x * BM;
  const int n0 = blockIdx.y * BN;
  const int tid = threadIdx.x;
  const int w = tid >> 5, l = tid & 31;
  const int g = l >> 4, lm = l & 15;
  const int mw = (w >> 2) * 32, nw = (w & 3) * 32;

  // per-thread staging assignment (3 x b128 per thread per tile)
  const int arow = tid >> 2, aco = (tid & 3) * 8;   // A: 64x32, 8 bf16/thread
  const int brow = tid >> 1, bco = (tid & 1) * 16;  // B: 128x32, 16 bf16/thread
  const bf16* agp = A + (size_t)(m0 + arow) * K + aco;
  const bf16* bgp = Bt + (size_t)(n0 + brow) * K + bco;
  const unsigned aoff[2] = {lds_off_of(&sA[0][arow * AS + aco]),
                            lds_off_of(&sA[1][arow * AS + aco])};
  const unsigned boff[2] = {lds_off_of(&sB[0][brow * BS + bco]),
                            lds_off_of(&sB[1][brow * BS + bco])};

  auto issue_tile = [&](int k0, int buf) {
    async_copy_b128(aoff[buf], agp + k0);
    async_copy_b128(boff[buf], bgp + k0);
    async_copy_b128(boff[buf] + 16, bgp + k0 + 8);
  };

  v8f acc[2][2] = {};
  const int nk = K / BK;
  issue_tile(0, 0);

  for (int it = 0; it < nk; ++it) {
    const int buf = it & 1;
    if (it + 1 < nk) {
      issue_tile((it + 1) * BK, buf ^ 1);
      asm volatile("s_wait_asynccnt 3" ::: "memory");  // this tile's 3 ops done
    } else {
      asm volatile("s_wait_asynccnt 0" ::: "memory");
    }
    __syncthreads();  // tile visible workgroup-wide

    v16bf a[2], bb[2];
#pragma unroll
    for (int mi = 0; mi < 2; mi++) {
      int row = mw + mi * 16 + lm;
      BF16V av;
      av.h[0] = *(const bf16x8*)&sA[buf][row * AS + g * 8];       // K=g*8..+7
      av.h[1] = *(const bf16x8*)&sA[buf][row * AS + 16 + g * 8];  // K=16+g*8..
      a[mi] = av.v;
    }
#pragma unroll
    for (int ni = 0; ni < 2; ni++) {
      int col = nw + ni * 16 + lm;
      BF16V bv;
      bv.h[0] = *(const bf16x8*)&sB[buf][col * BS + g * 8];       // K=g*8..+7
      bv.h[1] = *(const bf16x8*)&sB[buf][col * BS + 16 + g * 8];  // K=16+g*8..
      bb[ni] = bv.v;
    }
#pragma unroll
    for (int mi = 0; mi < 2; mi++)
#pragma unroll
      for (int ni = 0; ni < 2; ni++)
        acc[mi][ni] = __builtin_amdgcn_wmma_f32_16x16x32_bf16(
            false, a[mi], false, bb[ni], (short)0, acc[mi][ni], false, false);

    __syncthreads();  // all waves done reading buf before it is overwritten
  }

#pragma unroll
  for (int mi = 0; mi < 2; mi++)
#pragma unroll
    for (int ni = 0; ni < 2; ni++)
#pragma unroll
      for (int r = 0; r < 8; r++) {
        int row = m0 + mw + mi * 16 + (g << 3) + r;
        int col = n0 + nw + ni * 16 + lm;
        C[(size_t)row * N + col] = (OutT)acc[mi][ni][r];
      }
}

// ---------------------------------------------------------------------------
// Flash attention per (b,t,h). 4 waves; wave w owns query rows 16w..16w+15.
// q: (bt*64+i, h*64+d) bf16 ; kv row (bt*1024+j): cols[0:512)=k, [512:1024)=v.
// ---------------------------------------------------------------------------
__global__ __launch_bounds__(128) void attn_kernel(const bf16* __restrict__ q,
                                                   const bf16* __restrict__ kv,
                                                   bf16* __restrict__ out) {
  constexpr int Nl = 64, Nm = 1024, Dh = 64, inner = 512, J = 128;
  constexpr int QS = 72, KS = 72, VS = 136, PS = 136;
  __shared__ __align__(16) bf16 sQ[Nl * QS];
  __shared__ __align__(16) bf16 sK[J * KS];
  __shared__ __align__(16) bf16 sVt[Dh * VS];  // transposed: [d][j]
  __shared__ __align__(16) bf16 sP[Nl * PS];

  const int bt = blockIdx.x >> 3;
  const int h = blockIdx.x & 7;
  const int tid = threadIdx.x;
  const int w = tid >> 5, l = tid & 31, g = l >> 4, lm = l & 15;
  const float scale = 0.125f;  // Dh^-0.5

  {  // stage q (64x64) once
    const bf16* qsrc = q + (size_t)bt * Nl * inner + h * Dh;
    int row = tid >> 1, co = (tid & 1) * 32;
#pragma unroll
    for (int c = 0; c < 32; c += 8)
      *(bf16x8*)&sQ[row * QS + co + c] =
          *(const bf16x8*)(qsrc + (size_t)row * inner + co + c);
  }

  v8f acc_o[4] = {};
  float run_max[8], run_sum[8];
#pragma unroll
  for (int r = 0; r < 8; r++) { run_max[r] = -1e30f; run_sum[r] = 0.f; }

  const bf16* kbase = kv + (size_t)bt * Nm * 1024 + h * Dh;
  const bf16* vbase = kbase + inner;

  for (int j0 = 0; j0 < Nm; j0 += J) {
    __syncthreads();
    {  // stage k block row-major, v block transposed
      int j = tid;
      const bf16* ks = kbase + (size_t)(j0 + j) * 1024;
#pragma unroll
      for (int c = 0; c < 64; c += 8)
        *(bf16x8*)&sK[j * KS + c] = *(const bf16x8*)(ks + c);
      const bf16* vs = vbase + (size_t)(j0 + j) * 1024;
#pragma unroll
      for (int d = 0; d < 64; d++) sVt[d * VS + j] = vs[d];
    }
    __syncthreads();

    // s = q . k^T  (wave's 16 rows x 128 cols)
    v8f s[8] = {};
#pragma unroll
    for (int ks2 = 0; ks2 < 2; ks2++) {
      int row = w * 16 + lm;
      BF16V av;
      av.h[0] = *(const bf16x8*)&sQ[row * QS + ks2 * 32 + g * 8];
      av.h[1] = *(const bf16x8*)&sQ[row * QS + ks2 * 32 + 16 + g * 8];
#pragma unroll
      for (int ni = 0; ni < 8; ni++) {
        int col = ni * 16 + lm;
        BF16V bv;
        bv.h[0] = *(const bf16x8*)&sK[col * KS + ks2 * 32 + g * 8];
        bv.h[1] = *(const bf16x8*)&sK[col * KS + ks2 * 32 + 16 + g * 8];
        s[ni] = __builtin_amdgcn_wmma_f32_16x16x32_bf16(
            false, av.v, false, bv.v, (short)0, s[ni], false, false);
      }
    }

    // online softmax per row (row = w*16 + g*8 + r, stats shared by 16-lane half)
#pragma unroll
    for (int r = 0; r < 8; r++) {
      float m = -1e30f;
#pragma unroll
      for (int ni = 0; ni < 8; ni++) m = fmaxf(m, s[ni][r] * scale);
      for (int msk = 8; msk >= 1; msk >>= 1) m = fmaxf(m, __shfl_xor(m, msk, 32));
      float nmax = fmaxf(run_max[r], m);
      float corr = __expf(run_max[r] - nmax);
      run_max[r] = nmax;
      float ls = 0.f;
#pragma unroll
      for (int ni = 0; ni < 8; ni++) {
        float p = __expf(s[ni][r] * scale - nmax);
        ls += p;
        sP[(w * 16 + (g << 3) + r) * PS + ni * 16 + lm] = (bf16)p;
      }
      for (int msk = 8; msk >= 1; msk >>= 1) ls += __shfl_xor(ls, msk, 32);
      run_sum[r] = run_sum[r] * corr + ls;
#pragma unroll
      for (int di = 0; di < 4; di++) acc_o[di][r] *= corr;
    }

    // acc_o += p @ v  (K = 128 over j, only this wave's sP rows -> no barrier)
#pragma unroll
    for (int ks2 = 0; ks2 < 4; ks2++) {
      int row = w * 16 + lm;
      BF16V av;
      av.h[0] = *(const bf16x8*)&sP[row * PS + ks2 * 32 + g * 8];
      av.h[1] = *(const bf16x8*)&sP[row * PS + ks2 * 32 + 16 + g * 8];
#pragma unroll
      for (int di = 0; di < 4; di++) {
        int col = di * 16 + lm;
        BF16V bv;
        bv.h[0] = *(const bf16x8*)&sVt[col * VS + ks2 * 32 + g * 8];
        bv.h[1] = *(const bf16x8*)&sVt[col * VS + ks2 * 32 + 16 + g * 8];
        acc_o[di] = __builtin_amdgcn_wmma_f32_16x16x32_bf16(
            false, av.v, false, bv.v, (short)0, acc_o[di], false, false);
      }
    }
  }

  bf16* obase = out + (size_t)bt * Nl * inner + h * Dh;
#pragma unroll
  for (int di = 0; di < 4; di++)
#pragma unroll
    for (int r = 0; r < 8; r++) {
      int i = w * 16 + (g << 3) + r;
      int d = di * 16 + lm;
      obase[(size_t)i * inner + d] = (bf16)(acc_o[di][r] / run_sum[r]);
    }
}

// ---------------------------------------------------------------------------
extern "C" void kernel_launch(void* const* d_in, const int* in_sizes, int n_in,
                              void* d_out, int out_size, void* d_ws, size_t ws_size,
                              hipStream_t stream) {
  const float* x    = (const float*)d_in[0];  // (8,8,1024,1024)
  const float* lat  = (const float*)d_in[1];  // (8,8,64,2048)
  const float* nm_w = (const float*)d_in[2];
  const float* nm_b = (const float*)d_in[3];
  const float* nl_w = (const float*)d_in[4];
  const float* nl_b = (const float*)d_in[5];
  const float* Wq   = (const float*)d_in[6];  // (2048,512)
  const float* Wkv  = (const float*)d_in[7];  // (1024,1024)
  const float* Wout = (const float*)d_in[8];  // (512,2048)
  float* out = (float*)d_out;                 // (8,8,64,2048) f32

  char* ws = (char*)d_ws;
  bf16* xn    = (bf16*)(ws);                    // 65536 x 1024  (128 MB)
  bf16* kvb   = (bf16*)(ws + 134217728ull);     // 65536 x 1024  (128 MB)
  bf16* lnb   = (bf16*)(ws + 268435456ull);     //  4096 x 2048  ( 16 MB)
  bf16* qb    = (bf16*)(ws + 285212672ull);     //  4096 x  512  (  4 MB)
  bf16* ao    = (bf16*)(ws + 289406976ull);     //  4096 x  512  (  4 MB)
  bf16* WqT   = (bf16*)(ws + 293601280ull);     //   512 x 2048
  bf16* WkvT  = (bf16*)(ws + 295698432ull);     //  1024 x 1024
  bf16* WoutT = (bf16*)(ws + 297795584ull);     //  2048 x  512

  ln_kernel<1024><<<65536, 256, 0, stream>>>(x, nm_w, nm_b, xn);
  ln_kernel<2048><<<4096, 256, 0, stream>>>(lat, nl_w, nl_b, lnb);

  tconv_kernel<<<(2048 * 512) / 256, 256, 0, stream>>>(Wq, WqT, 2048, 512);
  tconv_kernel<<<(1024 * 1024) / 256, 256, 0, stream>>>(Wkv, WkvT, 1024, 1024);
  tconv_kernel<<<(512 * 2048) / 256, 256, 0, stream>>>(Wout, WoutT, 512, 2048);

  // q = ln @ Wq : M=4096 K=2048 N=512
  gemm_bf16<bf16><<<dim3(4096 / 64, 512 / 128), 256, 0, stream>>>(
      lnb, WqT, qb, 4096, 2048, 512);
  // kv = xn @ Wkv : M=65536 K=1024 N=1024  (dominant GEMM)
  gemm_bf16<bf16><<<dim3(65536 / 64, 1024 / 128), 256, 0, stream>>>(
      xn, WkvT, kvb, 65536, 1024, 1024);

  // attention over (b,t,h) = 512 workgroups
  attn_kernel<<<512, 128, 0, stream>>>(qb, kvb, ao);

  // out = attn_out @ Wout : M=4096 K=512 N=2048, f32 result
  gemm_bf16<float><<<dim3(4096 / 64, 2048 / 128), 256, 0, stream>>>(
      ao, WoutT, out, 4096, 512, 2048);
}
